// SimpleRNN_59476707115593
// MI455X (gfx1250) — matrix operand order repaired and checked by reading, not compiled, not run on hardware
//
#include <hip/hip_runtime.h>
#include <cmath>

// ---------------------------------------------------------------------------
// Persistent GRU-scan kernel for MI455X (gfx1250, wave32, WMMA).
// One block = 16 batch rows, all 512 timesteps.
//  - All weight B-fragments (W_ih, W_hh, head_W) are gathered from global
//    into per-lane REGISTERS once (loop-invariant), in WMMA B-fragment order.
//  - Recurrent state h lives in LDS as f32 (elementwise) and as f16
//    A-fragments (WMMA operand for the next step).
//  - Head projection is itself a WMMA (head_W zero-padded 8 -> 16 cols).
// ---------------------------------------------------------------------------

typedef __attribute__((ext_vector_type(16))) _Float16 v16h;
typedef __attribute__((ext_vector_type(8)))  float    v8f;

constexpr int B_    = 1024;
constexpr int K_    = 512;
constexpr int U_    = 3;
constexpr int P_    = 5;
constexpr int LIFT_ = 32;
constexpr int H_    = 128;
constexpr int G3_   = 3 * H_;       // 384 gate columns
constexpr int ROWS  = 16;           // batch rows per block (= WMMA M)
constexpr int THREADS = 256;        // 8 waves
constexpr int BLOCKS  = B_ / ROWS;  // 64

// ---- dynamic LDS layout (bytes); WMMA-fragment regions kept 32B aligned ----
constexpr size_t OFF_GI    = 0;                          // [24][32][8] f32 (frag order)
constexpr size_t OFF_GH    = OFF_GI    + 24*32*8*4;      // [24][32][8] f32
constexpr size_t OFF_HS    = OFF_GH    + 24*32*8*4;      // [16][128] f32
constexpr size_t OFF_HFRAG = OFF_HS    + 16*128*4;       // [4][32][16] f16 A-frags of h
constexpr size_t OFF_XFRAG = OFF_HFRAG + 4*32*16*2;      // [32][16] f16 A-frag of x
constexpr size_t OFF_LW    = OFF_XFRAG + 32*16*2;        // [32][8] f32
constexpr size_t OFF_LB    = OFF_LW    + 32*8*4;         // [32] f32
constexpr size_t OFF_U2Y   = OFF_LB    + 32*4;           // [3][5] f32 (pad)
constexpr size_t OFF_YS    = OFF_U2Y   + 16*4;           // [16][5] f32 (pad)
constexpr size_t OFF_TH    = OFF_YS    + 96*4;           // [16][8] f32
constexpr size_t OFF_US    = OFF_TH    + 16*8*4;         // [16][3] f32 (pad)
constexpr size_t OFF_DT    = OFF_US    + 64*4;           // [16] f32
constexpr size_t SMEM_BYTES= OFF_DT    + 16*4;           // ~63.4 KB

__device__ __forceinline__ float sigm_(float x) { return 1.0f / (1.0f + __expf(-x)); }

// frag index for gi/gh stores/loads: column C (0..383), row m -> (g = m>>3, v = m&7)
__device__ __forceinline__ int frag_idx(int C, int g, int v) {
    int n0 = C >> 4, n = C & 15;
    return (n0 * 32 + g * 16 + n) * 8 + v;
}

// A-fragment (16x32 f16) slot for element (m, kk): lane = g*16+m, element e
__device__ __forceinline__ void afrag_slot(int kk, int m, int& lane, int& e) {
    int g = (kk >> 3) & 1;
    e = (kk & 7) + ((kk >= 16) ? 8 : 0);
    lane = g * 16 + m;
}

__device__ __forceinline__ void rhs5(const float* y, const float* th, float* d) {
    float A = y[0], D = y[1], G = y[2], J = y[3], M = y[4];
    float kf1 = th[0], kf2 = th[1], kf3 = th[2], kf4 = th[3];
    float kr1 = th[4], kr2 = th[5], kr3 = th[6], kr4 = th[7];
    d[0] = -kf1 * A + kr1 * D;
    d[1] =  kf1 * A - kr1 * D - kf2 * D + kr2 * G;
    d[2] =  kf2 * D - kr2 * G - kf3 * G + kr3 * J;
    d[3] =  kf3 * G - kr3 * J - kf4 * J + kr4 * M;
    d[4] =  kf4 * J - kr4 * M;
}

__global__ __launch_bounds__(THREADS)
void gru_scan_wmma_kernel(const float* __restrict__ y0,
                          const float* __restrict__ u_seq,
                          const float* __restrict__ dt_seq,
                          const float* __restrict__ lift_W,
                          const float* __restrict__ lift_b,
                          const float* __restrict__ W_ih,
                          const float* __restrict__ W_hh,
                          const float* __restrict__ b_ih,
                          const float* __restrict__ b_hh,
                          const float* __restrict__ head_W,
                          const float* __restrict__ head_b,
                          const float* __restrict__ u_to_y,
                          float* __restrict__ out) {
    extern __shared__ char smem[];
    float*    sGi    = (float*)   (smem + OFF_GI);
    float*    sGh    = (float*)   (smem + OFF_GH);
    float*    sH     = (float*)   (smem + OFF_HS);
    _Float16* sHfrag = (_Float16*)(smem + OFF_HFRAG);
    _Float16* sXfrag = (_Float16*)(smem + OFF_XFRAG);
    float*    sLw    = (float*)   (smem + OFF_LW);
    float*    sLb    = (float*)   (smem + OFF_LB);
    float*    sU2y   = (float*)   (smem + OFF_U2Y);
    float*    sY     = (float*)   (smem + OFF_YS);
    float*    sTh    = (float*)   (smem + OFF_TH);
    float*    sU     = (float*)   (smem + OFF_US);
    float*    sDt    = (float*)   (smem + OFF_DT);

    const int tid  = threadIdx.x;
    const int lane = tid & 31;
    const int wave = tid >> 5;
    const int row0 = blockIdx.x * ROWS;

    // -------- one-time: gather loop-invariant B-fragments into registers ----
    // B-fragment layout (K=32 chunk): n = lane&15, element e -> k = (e&7) + 8*(lane>>4) + 16*(e>>3).
    v16h rBI[3];        // W_ih^T   tiles, K = LIFT = 32 (one chunk)
    v16h rBH[3][4];     // W_hh^T   tiles, 4 K-chunks of 32
    v16h rBHead[4];     // head_W^T padded to 16 cols, 4 K-chunks of 32
    float biasI[3], biasH[3];
    #pragma unroll
    for (int t = 0; t < 3; ++t) {
        const int n0 = wave * 3 + t;
        v16h bi;
        #pragma unroll
        for (int e = 0; e < 16; ++e)
            bi[e] = (_Float16)W_ih[(n0 * 16 + (lane & 15)) * LIFT_ + (e & 7) + ((lane >> 4) << 3) + ((e >> 3) << 4)];
        rBI[t] = bi;
        #pragma unroll
        for (int kc = 0; kc < 4; ++kc) {
            v16h bh;
            #pragma unroll
            for (int e = 0; e < 16; ++e)
                bh[e] = (_Float16)W_hh[(n0 * 16 + (lane & 15)) * H_ + kc * 32 + (e & 7) + ((lane >> 4) << 3) + ((e >> 3) << 4)];
            rBH[t][kc] = bh;
        }
        biasI[t] = b_ih[n0 * 16 + (lane & 15)];
        biasH[t] = b_hh[n0 * 16 + (lane & 15)];
    }
    #pragma unroll
    for (int kc = 0; kc < 4; ++kc) {
        v16h bh;
        #pragma unroll
        for (int e = 0; e < 16; ++e)
            bh[e] = ((lane & 15) < 8) ? (_Float16)head_W[(lane & 15) * H_ + kc * 32 + (e & 7) + ((lane >> 4) << 3) + ((e >> 3) << 4)]
                            : (_Float16)0.0f;
        rBHead[kc] = bh;
    }
    const float hbias = ((lane & 15) < 8) ? head_b[lane & 15] : 0.0f;

    // -------- one-time: LDS init --------------------------------------------
    for (int i = tid; i < LIFT_ * (U_ + P_); i += THREADS) sLw[i] = lift_W[i];
    if (tid < LIFT_)     sLb[tid]  = lift_b[tid];
    if (tid < U_ * P_)   sU2y[tid] = u_to_y[tid];
    if (tid < ROWS * P_) sY[tid]   = y0[(size_t)(row0 + tid / P_) * P_ + tid % P_] + 0.01f;
    for (int i = tid; i < ROWS * H_;   i += THREADS) sH[i] = 0.0f;
    for (int i = tid; i < 4 * 32 * 16; i += THREADS) sHfrag[i] = (_Float16)0.0f;
    __syncthreads();

    float* yout  = out;                          // (B,K,P) flat
    float* thout = out + (size_t)B_ * K_ * P_;   // (B,K,8) flat

    for (int k = 0; k < K_; ++k) {
        // -------- phase 1: stage u_k, dt_k --------------------------------
        if (tid < ROWS * U_) {
            int r = tid / U_, c = tid % U_;
            const float* up = &u_seq[(size_t)(row0 + r) * K_ * U_ + (size_t)k * U_ + c];
            sU[tid] = *up;
            if (k + 1 < K_) __builtin_prefetch(up + U_, 0, 0);   // global_prefetch
        }
        if (tid < ROWS) sDt[tid] = dt_seq[(size_t)(row0 + tid) * K_ + k];
        __syncthreads();

        // -------- phase 2: lift + silu -> x A-fragment (f16) --------------
        {
            int r = tid & 15, c = tid >> 4;                  // c in 0..15
            for (int cc = c; cc < LIFT_; cc += 16) {
                float acc = sLb[cc];
                #pragma unroll
                for (int j = 0; j < U_; ++j) acc += sU[r * U_ + j] * sLw[cc * 8 + j];
                #pragma unroll
                for (int j = 0; j < P_; ++j) acc += sY[r * P_ + j] * sLw[cc * 8 + U_ + j];
                float x = acc * sigm_(acc);                  // silu
                int ln, e; afrag_slot(cc, r, ln, e);
                sXfrag[ln * 16 + e] = (_Float16)x;
            }
        }
        __syncthreads();

        // -------- phase 3: WMMA — gi and gh, 3 N-tiles per wave -----------
        {
            v16h ax = *(const v16h*)(sXfrag + lane * 16);
            v16h ah[4];
            #pragma unroll
            for (int kc = 0; kc < 4; ++kc)
                ah[kc] = *(const v16h*)(sHfrag + (kc * 32 + lane) * 16);

            #pragma unroll
            for (int t = 0; t < 3; ++t) {
                const int n0 = wave * 3 + t;
                const float bi = biasI[t], bh = biasH[t];
                v8f accI = {bi, bi, bi, bi, bi, bi, bi, bi};
                v8f accH = {bh, bh, bh, bh, bh, bh, bh, bh};

                accI = __builtin_amdgcn_wmma_f32_16x16x32_f16(
                        false, ax, false, rBI[t], (short)0, accI, false, false);
                #pragma unroll
                for (int kc = 0; kc < 4; ++kc)
                    accH = __builtin_amdgcn_wmma_f32_16x16x32_f16(
                            false, ah[kc], false, rBH[t][kc], (short)0, accH,
                            false, false);

                *(v8f*)(sGi + (n0 * 32 + lane) * 8) = accI;
                *(v8f*)(sGh + (n0 * 32 + lane) * 8) = accH;
            }
        }
        __syncthreads();

        // -------- phase 4: GRU gates (8 cols per thread) ------------------
        {
            int m = tid >> 4, c0 = (tid & 15) * 8;
            int g = m >> 3, v = m & 7;
            #pragma unroll
            for (int jj = 0; jj < 8; ++jj) {
                int j = c0 + jj;
                float ir = sGi[frag_idx(j,          g, v)];
                float iz = sGi[frag_idx(H_ + j,     g, v)];
                float in_= sGi[frag_idx(2 * H_ + j, g, v)];
                float hr = sGh[frag_idx(j,          g, v)];
                float hz = sGh[frag_idx(H_ + j,     g, v)];
                float hn = sGh[frag_idx(2 * H_ + j, g, v)];
                float r  = sigm_(ir + hr);
                float z  = sigm_(iz + hz);
                float n  = tanhf(in_ + r * hn);
                float ho = sH[m * H_ + j];
                float hnew = (1.0f - z) * n + z * ho;
                sH[m * H_ + j] = hnew;
                int ln, e; afrag_slot(j & 31, m, ln, e);
                sHfrag[((j >> 5) * 32 + ln) * 16 + e] = (_Float16)hnew;
            }
        }
        __syncthreads();

        // -------- phase 5: head projection via WMMA (wave 0) --------------
        if (wave == 0) {
            v16h ahn[4];
            #pragma unroll
            for (int kc = 0; kc < 4; ++kc)
                ahn[kc] = *(const v16h*)(sHfrag + (kc * 32 + lane) * 16);
            v8f acc = {hbias, hbias, hbias, hbias, hbias, hbias, hbias, hbias};
            #pragma unroll
            for (int kc = 0; kc < 4; ++kc)
                acc = __builtin_amdgcn_wmma_f32_16x16x32_f16(
                        false, ahn[kc], false, rBHead[kc], (short)0, acc,
                        false, false);
            const int n = lane & 15, g = lane >> 4;
            if (n < 8) {
                #pragma unroll
                for (int v = 0; v < 8; ++v) {
                    int m = v + 8 * g;
                    float th = 0.001f + (2.0f - 0.001f) * sigm_(acc[v]);
                    sTh[m * 8 + n] = th;
                    thout[(size_t)(row0 + m) * K_ * 8 + (size_t)k * 8 + n] = th;
                }
            }
        }
        __syncthreads();

        // -------- phase 6: RK4 ODE step per batch row ---------------------
        if (tid < ROWS) {
            int m = tid;
            float th[8];
            #pragma unroll
            for (int i = 0; i < 8; ++i) th[i] = sTh[m * 8 + i];
            float y[5];
            #pragma unroll
            for (int p = 0; p < P_; ++p) {
                float yj = sY[m * P_ + p];
                #pragma unroll
                for (int uu = 0; uu < U_; ++uu)
                    yj += sU[m * U_ + uu] * sU2y[uu * P_ + p];
                y[p] = yj;
            }
            float dt = sDt[m];
            float k1[5], k2[5], k3[5], k4[5], t1[5];
            rhs5(y, th, k1);
            #pragma unroll
            for (int p = 0; p < 5; ++p) t1[p] = y[p] + 0.5f * dt * k1[p];
            rhs5(t1, th, k2);
            #pragma unroll
            for (int p = 0; p < 5; ++p) t1[p] = y[p] + 0.5f * dt * k2[p];
            rhs5(t1, th, k3);
            #pragma unroll
            for (int p = 0; p < 5; ++p) t1[p] = y[p] + dt * k3[p];
            rhs5(t1, th, k4);
            #pragma unroll
            for (int p = 0; p < 5; ++p) {
                float yn = y[p] + dt * (1.0f / 6.0f) *
                           (k1[p] + 2.0f * k2[p] + 2.0f * k3[p] + k4[p]);
                yn = fmaxf(yn, 0.0f);
                sY[m * P_ + p] = yn;
                yout[(size_t)(row0 + m) * K_ * P_ + (size_t)k * P_ + p] = yn;
            }
        }
        __syncthreads();
    }
}

extern "C" void kernel_launch(void* const* d_in, const int* in_sizes, int n_in,
                              void* d_out, int out_size, void* d_ws, size_t ws_size,
                              hipStream_t stream) {
    const float* y0     = (const float*)d_in[0];
    const float* u_seq  = (const float*)d_in[1];
    const float* dt_seq = (const float*)d_in[2];
    const float* lift_W = (const float*)d_in[3];
    const float* lift_b = (const float*)d_in[4];
    const float* W_ih   = (const float*)d_in[5];
    const float* W_hh   = (const float*)d_in[6];
    const float* b_ih   = (const float*)d_in[7];
    const float* b_hh   = (const float*)d_in[8];
    const float* head_W = (const float*)d_in[9];
    const float* head_b = (const float*)d_in[10];
    const float* u_to_y = (const float*)d_in[11];
    float* out = (float*)d_out;

    (void)in_sizes; (void)n_in; (void)out_size; (void)d_ws; (void)ws_size;

    hipFuncSetAttribute(reinterpret_cast<const void*>(gru_scan_wmma_kernel),
                        hipFuncAttributeMaxDynamicSharedMemorySize,
                        (int)SMEM_BYTES);

    gru_scan_wmma_kernel<<<BLOCKS, THREADS, SMEM_BYTES, stream>>>(
        y0, u_seq, dt_seq, lift_W, lift_b, W_ih, W_hh, b_ih, b_hh,
        head_W, head_b, u_to_y, out);
}
